// MemN2NBlock_21938692948089
// MI455X (gfx1250) — matrix operand order repaired and checked by reading, not compiled, not run on hardware
//
#include <hip/hip_runtime.h>

// ---------------------------------------------------------------------------
// MemN2N fused block for gfx1250 (MI455X), bf16 WMMA, f32 accumulate.
//   scores = story_a (B,M,D) . u^T (B,D,Q)   -> softmax over Q per (b,m)
//   c      = P^T (B,Q,M) . story_c (B,M,D)
//   out    = c + u.H
// B=32, M=8192, Q=64, D=128. Bandwidth-bound: ~270MB @ 23.3TB/s.
// story_c tile staging uses GLOBAL_LOAD_ASYNC_TO_LDS_B128 (ASYNCcnt); the DMA
// overlaps the score WMMAs, waited with s_wait_asynccnt before the barrier.
// ---------------------------------------------------------------------------

typedef __attribute__((ext_vector_type(16))) __bf16 v16bf;
typedef __attribute__((ext_vector_type(8)))  float  v8f;

#define B_ 32
#define M_ 8192
#define Q_ 64
#define D_ 128

// LDS strides (in dwords / shorts), padded for bank-conflict freedom.
#define SS_STRIDE 68        // S[m][q]: 32 rows x 64 f32, 272B rows (16B aligned)
#define SC_STRIDE 132       // story_c tile: 32 rows x 128 f32, 528B rows
#define SP_STRIDE_S 40      // P^T: 64 q-rows x 32 m bf16, 80B rows (16B aligned)

#if defined(__gfx1250__) && __has_builtin(__builtin_amdgcn_global_load_async_to_lds_b128)
#define USE_ASYNC_LDS 1
#else
#define USE_ASYNC_LDS 0
#endif

// b128 async copy operand types: pointer to 16-byte int vector, AS-qualified.
typedef int v4i_ __attribute__((vector_size(16)));
typedef __attribute__((address_space(1))) v4i_ gv4i;   // global
typedef __attribute__((address_space(3))) v4i_ lv4i;   // LDS

static __device__ __forceinline__ void wait_async0() {
#if __has_builtin(__builtin_amdgcn_s_wait_asynccnt)
    __builtin_amdgcn_s_wait_asynccnt(0);
#else
    asm volatile("s_wait_asynccnt 0x0" ::: "memory");
#endif
}

union Bf16Frag {
    v16bf v;
    uint4 q[2];
};

static __device__ __forceinline__ unsigned short f2bf_bits(float x) {
    union { __bf16 h; unsigned short s; } cv;
    cv.h = (__bf16)x;
    return cv.s;
}

__global__ __launch_bounds__(256)
void memn2n_attend(const float* __restrict__ story_a,
                   const float* __restrict__ u,
                   const float* __restrict__ story_c,
                   float* __restrict__ ws,
                   int chunks, int mchunk) {
    __shared__ __align__(16) float          sS[32 * SS_STRIDE];
    __shared__ __align__(16) float          sC[32 * SC_STRIDE];
    __shared__ __align__(16) unsigned short sP[Q_ * SP_STRIDE_S];

    const int tid  = threadIdx.x;
    const int wave = tid >> 5;       // 0..7
    const int lane = tid & 31;
    const int lh   = lane >> 4;      // lane half (0/1)
    const int ln   = lane & 15;

    const int b     = blockIdx.x / chunks;
    const int chunk = blockIdx.x % chunks;
    const int m0    = chunk * mchunk;

    const int mt = wave >> 2;        // score m-subtile (0/1) within 32-row tile
    const int qt = wave & 3;         // score q-tile (0..3)
    // second GEMM: wave owns D columns [wave*16, wave*16+16)

    // --- U^T B-fragments, resident in VGPRs for the whole M loop.
    // B layout (32x16 bf16): lane -> col n=ln, k = lh*16 + j (pairs packed).
    v16bf ubf[4];
    {
        const float* ub = u + ((size_t)b * Q_ + (size_t)qt * 16 + ln) * D_;
        #pragma unroll
        for (int ks = 0; ks < 4; ++ks) {
            const float* p = ub + ks * 32 + lh * 16;
            v16bf f;
            #pragma unroll
            for (int j = 0; j < 16; ++j) f[j] = (__bf16)p[j];
            ubf[ks] = f;
        }
    }

    v8f acc[4] = {};                 // C[q-tile][this wave's 16 d-cols]

    const int iters = mchunk / 32;
    // story_a A-fragment row for this lane (A layout: row = ln for both halves)
    const float* sa_row = story_a + ((size_t)b * M_ + m0 + mt * 16 + ln) * D_;
    // cooperative story_c staging: 256 threads cover 32 rows x 512B
    const int crow = tid >> 3, cseg = tid & 7;
    const float* sc_row = story_c + ((size_t)b * M_ + m0 + crow) * D_ + cseg * 16;

    for (int it = 0; it < iters; ++it) {
        __syncthreads();   // previous iteration done reading sC/sP

        // ---- stage story_c 32x128 f32 tile into LDS (coalesced, 64B/thread)
        {
            const float* g = sc_row + (size_t)it * 32 * D_;
            float*     dst = &sC[crow * SC_STRIDE + cseg * 16];
#if USE_ASYNC_LDS
            // Async DMA global->LDS; lands before the s_wait_asynccnt below.
            // AS1 value == flat value for global; AS3 offset == low 32 flat bits.
            gv4i* gp = (gv4i*)(unsigned long long)(size_t)(const void*)g;
            lv4i* lp = (lv4i*)(unsigned int)(size_t)(void*)dst;
            __builtin_amdgcn_global_load_async_to_lds_b128(gp, lp, 0, 0);
            __builtin_amdgcn_global_load_async_to_lds_b128(gp, lp, 16, 0);
            __builtin_amdgcn_global_load_async_to_lds_b128(gp, lp, 32, 0);
            __builtin_amdgcn_global_load_async_to_lds_b128(gp, lp, 48, 0);
#else
            const float4* g4 = (const float4*)g;
            float4 a0 = g4[0], a1 = g4[1], a2 = g4[2], a3 = g4[3];
            *(float4*)(dst + 0)  = a0;
            *(float4*)(dst + 4)  = a1;
            *(float4*)(dst + 8)  = a2;
            *(float4*)(dst + 12) = a3;
#endif
        }
        if (it + 1 < iters) {
            __builtin_prefetch(sa_row + (size_t)(it + 1) * 32 * D_, 0, 1);
            __builtin_prefetch(sc_row + (size_t)(it + 1) * 32 * D_, 0, 1);
        }

        // ---- score tile: S(16x16) = A(16x128) x U^T(128x16), 4 k-slices of 32
        v8f s = {};
        {
            const float* ap = sa_row + (size_t)it * 32 * D_;
            #pragma unroll
            for (int ks = 0; ks < 4; ++ks) {
                // A layout (16x32 bf16): lane row=ln; k = lh*16+j
                const float* p0 = ap + ks * 32 + lh * 16;
                v16bf af;
                #pragma unroll
                for (int j = 0; j < 8; ++j) af[j] = (__bf16)p0[j];
                #pragma unroll
                for (int j = 0; j < 8; ++j) af[8 + j] = (__bf16)p0[8 + j];
                s = __builtin_amdgcn_wmma_f32_16x16x32_bf16(
                        false, af, false, ubf[ks], (short)0, s, false, false);
            }
        }
        // C/D layout: vgpr r, lane -> row r+8*lh, col ln
        #pragma unroll
        for (int r = 0; r < 8; ++r)
            sS[(mt * 16 + r + 8 * lh) * SS_STRIDE + qt * 16 + ln] = s[r];

#if USE_ASYNC_LDS
        wait_async0();   // sC tile resident before we publish via the barrier
#endif
        __syncthreads();

        // ---- softmax over q=64 per m-row; 8 threads per row (shfl butterfly)
        {
            const int row = tid >> 3, j = tid & 7;
            float* rp = &sS[row * SS_STRIDE + j * 8];
            float4 v0 = *(float4*)rp, v1 = *(float4*)(rp + 4);
            float e[8] = { v0.x, v0.y, v0.z, v0.w, v1.x, v1.y, v1.z, v1.w };
            float mx = e[0];
            #pragma unroll
            for (int i = 1; i < 8; ++i) mx = fmaxf(mx, e[i]);
            mx = fmaxf(mx, __shfl_xor(mx, 1, 32));
            mx = fmaxf(mx, __shfl_xor(mx, 2, 32));
            mx = fmaxf(mx, __shfl_xor(mx, 4, 32));
            float sum = 0.f;
            #pragma unroll
            for (int i = 0; i < 8; ++i) { e[i] = __expf(e[i] - mx); sum += e[i]; }
            sum += __shfl_xor(sum, 1, 32);
            sum += __shfl_xor(sum, 2, 32);
            sum += __shfl_xor(sum, 4, 32);
            float inv = 1.0f / sum;
            // write P^T[q][m] as bf16 (row stride 80B -> conflict-free, 16B aligned)
            #pragma unroll
            for (int i = 0; i < 8; ++i)
                sP[(j * 8 + i) * SP_STRIDE_S + row] = f2bf_bits(e[i] * inv);
        }
        __syncthreads();

        // ---- second GEMM: acc[q2] += P^T(16x32) x story_c_sub(32x16)
        // B fragment from staged sC: col n = wave*16+ln, k = lh*16 + j
        v16bf cb;
        #pragma unroll
        for (int j = 0; j < 16; ++j)
            cb[j] = (__bf16)sC[(lh * 16 + j) * SC_STRIDE + wave * 16 + ln];

        #pragma unroll
        for (int q2 = 0; q2 < 4; ++q2) {
            // A fragment straight from sP (contiguous k = lh*16 + j, same order as cb)
            const char* base = (const char*)sP + (q2 * 16 + ln) * (SP_STRIDE_S * 2) + lh * 32;
            Bf16Frag ap;
            ap.q[0] = *(const uint4*)(base);
            ap.q[1] = *(const uint4*)(base + 16);
            acc[q2] = __builtin_amdgcn_wmma_f32_16x16x32_bf16(
                          false, ap.v, false, cb, (short)0, acc[q2], false, false);
        }
    }

    // ---- write this chunk's partial C (deterministic; reduced by kernel 2)
    float* wsp = ws + (size_t)blockIdx.x * (Q_ * D_);
    #pragma unroll
    for (int q2 = 0; q2 < 4; ++q2) {
        #pragma unroll
        for (int r = 0; r < 8; ++r) {
            const int q = q2 * 16 + r + 8 * lh;
            const int d = wave * 16 + ln;
            wsp[q * D_ + d] = acc[q2][r];
        }
    }
}

// out[b,q,d] = sum_chunks partial[b,chunk,q,d] + sum_k u[b,q,k]*H[k,d]
__global__ __launch_bounds__(128)
void memn2n_combine(const float* __restrict__ u, const float* __restrict__ H,
                    const float* __restrict__ ws, float* __restrict__ out,
                    int chunks) {
    const int bq = blockIdx.x;           // 0 .. B*Q-1
    const int b = bq / Q_, q = bq % Q_;
    const int d = threadIdx.x;
    const float* urow = u + ((size_t)b * Q_ + q) * D_;
    float acc = 0.f;
    #pragma unroll 8
    for (int k = 0; k < D_; ++k) acc += urow[k] * H[k * D_ + d];  // H rows coalesced
    for (int c = 0; c < chunks; ++c)
        acc += ws[(((size_t)(b * chunks + c)) * Q_ + q) * D_ + d];
    out[((size_t)b * Q_ + q) * D_ + d] = acc;
}

extern "C" void kernel_launch(void* const* d_in, const int* in_sizes, int n_in,
                              void* d_out, int out_size, void* d_ws, size_t ws_size,
                              hipStream_t stream) {
    const float* story_a = (const float*)d_in[0];
    const float* u       = (const float*)d_in[1];
    const float* story_c = (const float*)d_in[2];
    const float* H       = (const float*)d_in[3];
    float* out = (float*)d_out;
    float* ws  = (float*)d_ws;

    // One partial (B,Q,D) set per M-chunk: 1 MB each. Pick largest pow2 <= 8
    // that fits the provided workspace (deterministic per run).
    const size_t per = (size_t)B_ * Q_ * D_ * sizeof(float);
    size_t fit = ws_size / per;
    int chunks = 1;
    while (chunks < 8 && (size_t)(chunks * 2) <= fit) chunks *= 2;
    const int mchunk = M_ / chunks;

    memn2n_attend<<<dim3(B_ * chunks), dim3(256), 0, stream>>>(
        story_a, u, story_c, ws, chunks, mchunk);
    memn2n_combine<<<dim3(B_ * Q_), dim3(128), 0, stream>>>(
        u, H, ws, out, chunks);
}
